// SMILModel_11768210391043
// MI455X (gfx1250) — matrix operand order repaired and checked by reading, not compiled, not run on hardware
//
#include <hip/hip_runtime.h>
#include <hip/hip_bf16.h>
#include <stdint.h>

// ---------------------------------------------------------------------------
// Types for CDNA5 WMMA (wave32, v_wmma_f32_16x16x32_bf16)
// ---------------------------------------------------------------------------
typedef __attribute__((ext_vector_type(16))) __bf16        bf16x16;
typedef __attribute__((ext_vector_type(8)))  float         f32x8;
typedef __attribute__((ext_vector_type(4)))  unsigned int  u32x4;
typedef __attribute__((ext_vector_type(4)))  float         f32x4;
typedef unsigned short ushort_t;

#define BMT 256   // block tile M
#define BNT 128   // block tile N
#define BK  32    // K step (= WMMA K)
#define ROWB 80   // bytes per LDS row: 64B data (32 bf16) + 16B pad (16B aligned)

// ---------------------------------------------------------------------------
// CDNA5 async global->LDS copy (ASYNCcnt), guarded; fallback = plain ld/st.
// Probe-confirmed signature: (v4i AS1* src, v4i AS3* dst, imm offset, imm cpol)
// ---------------------------------------------------------------------------
#if defined(__has_builtin)
# if __has_builtin(__builtin_amdgcn_global_load_async_to_lds_b128)
#  define HAVE_ASYNC_LDS 1
# endif
# if __has_builtin(__builtin_amdgcn_s_wait_asynccnt)
#  define HAVE_WAIT_ASYNC 1
# endif
#endif
#ifndef HAVE_ASYNC_LDS
# define HAVE_ASYNC_LDS 0
#endif
#ifndef HAVE_WAIT_ASYNC
# define HAVE_WAIT_ASYNC 0
#endif

#if HAVE_ASYNC_LDS
typedef int v4i_g __attribute__((vector_size(16)));
typedef __attribute__((address_space(1))) v4i_g* as1_v4i_ptr;
typedef __attribute__((address_space(3))) v4i_g* as3_v4i_ptr;
#endif

__device__ __forceinline__ void async_cp16(void* ldst, const void* gsrc) {
#if HAVE_ASYNC_LDS
  __builtin_amdgcn_global_load_async_to_lds_b128(
      (as1_v4i_ptr)gsrc, (as3_v4i_ptr)ldst, 0, 0);
#else
  *(u32x4*)ldst = *(const u32x4*)gsrc;          // global_load_b128 + ds_store_b128
#endif
}

template<int NWAIT>
__device__ __forceinline__ void wait_async() {
#if HAVE_ASYNC_LDS
# if HAVE_WAIT_ASYNC
  __builtin_amdgcn_s_wait_asynccnt(NWAIT);
# else
  asm volatile("s_wait_asynccnt %0" :: "i"(NWAIT) : "memory");
# endif
#endif
}

__device__ __forceinline__ unsigned short f2bf(float f) {
  union { float f; unsigned u; } v; v.f = f;
  unsigned r = v.u + 0x7FFFu + ((v.u >> 16) & 1u);   // round-to-nearest-even
  return (unsigned short)(r >> 16);
}
__device__ __forceinline__ unsigned pack2(float lo, float hi) {
  return (unsigned)f2bf(lo) | ((unsigned)f2bf(hi) << 16);
}

union FragCvt { u32x4 u[2]; bf16x16 v; };
__device__ __forceinline__ bf16x16 ld_frag(const unsigned char* p0, const unsigned char* p1) {
  FragCvt f;
  f.u[0] = *(const u32x4*)p0;   // ds_load_b128
  f.u[1] = *(const u32x4*)p1;   // ds_load_b128
  return f.v;
}

constexpr int EPI_NONE = 0;
constexpr int EPI_RELU = 1;
constexpr int EPI_EXPW = 2;   // Cf=exp(v); C2f=1+sqrt(Cf)*eps; Cb=bf16(C2f)

// ---------------------------------------------------------------------------
// bf16 WMMA GEMM, double-buffered async LDS staging.
//   C = A[M,K](bf16) * op(B)(bf16) + bias, epilogue templated.
//   TRANSB=1: B is [N,K] (K contiguous, raw async copy)
//   TRANSB=0: B is [K,N] (transposed while staging, sync path)
// 256 threads = 8 waves; wave (wm=wid>>1, wn=wid&1) owns 64x64 = 4x4 WMMA tiles.
// ---------------------------------------------------------------------------
template<int EPI, bool TRANSB>
__global__ __launch_bounds__(256)
void gemm_bf16(const ushort_t* __restrict__ A, const ushort_t* __restrict__ Bm,
               const float* __restrict__ bias, const float* __restrict__ eps,
               float* __restrict__ Cf, float* __restrict__ C2f, ushort_t* __restrict__ Cb,
               int M, int N, int K)
{
  __shared__ unsigned char lds[2 * (BMT + BNT) * ROWB];   // 60 KB, 2 stages

  const int t    = threadIdx.x;
  const int lane = t & 31;
  const int wid  = t >> 5;
  const int wm   = wid >> 1;       // 0..3 : 64-row strip
  const int wn   = wid & 1;        // 0..1 : 64-col strip
  const bool hi  = lane >= 16;
  const int m0   = blockIdx.y * BMT;
  const int n0   = blockIdx.x * BNT;

  f32x8 acc[4][4];
  #pragma unroll
  for (int i = 0; i < 4; ++i)
    #pragma unroll
    for (int j = 0; j < 4; ++j) acc[i][j] = 0.0f;

  auto stageA = [&](int s) -> unsigned char* { return lds + s * (BMT + BNT) * ROWB; };
  auto stageB = [&](int s) -> unsigned char* { return stageA(s) + BMT * ROWB; };

  auto copy_stage = [&](int s, int k0) {
    unsigned char* la = stageA(s);
    #pragma unroll
    for (int i = 0; i < 4; ++i) {                 // A: 1024 x 16B chunks
      int e = t + i * 256;
      int row = e >> 2, c = e & 3;
      async_cp16(la + row * ROWB + c * 16, A + (size_t)(m0 + row) * K + k0 + c * 8);
    }
    unsigned char* lb = stageB(s);
    if (TRANSB) {
      #pragma unroll
      for (int i = 0; i < 2; ++i) {               // B: 512 x 16B chunks
        int e = t + i * 256;
        int row = e >> 2, c = e & 3;
        int n = n0 + row;
        if (n < N)
          async_cp16(lb + row * ROWB + c * 16, Bm + (size_t)n * K + k0 + c * 8);
      }
    } else {
      ushort_t* lb16 = (ushort_t*)lb;             // transpose [K][N] -> [n][k]
      #pragma unroll
      for (int i = 0; i < 4; ++i) {               // 1024 groups of 4 along N
        int e = t + i * 256;
        int k = e >> 5, ns = e & 31;
        int n = n0 + ns * 4;
        unsigned v01 = 0, v23 = 0;
        const ushort_t* src = Bm + (size_t)(k0 + k) * N + n;
        if (n < N) {
          const unsigned* s2 = (const unsigned*)src;
          v01 = s2[0]; v23 = s2[1];
          __builtin_prefetch(src + (size_t)BK * N, 0, 0);   // global_prefetch_b8
        }
        lb16[(ns * 4 + 0) * (ROWB / 2) + k] = (ushort_t)v01;
        lb16[(ns * 4 + 1) * (ROWB / 2) + k] = (ushort_t)(v01 >> 16);
        lb16[(ns * 4 + 2) * (ROWB / 2) + k] = (ushort_t)v23;
        lb16[(ns * 4 + 3) * (ROWB / 2) + k] = (ushort_t)(v23 >> 16);
      }
    }
  };

  const int nk = K / BK;
  copy_stage(0, 0);
  for (int kt = 0; kt < nk; ++kt) {
    if (kt + 1 < nk) {
      copy_stage((kt + 1) & 1, (kt + 1) * BK);    // overlap next stage
      wait_async<TRANSB ? 6 : 4>();               // stage kt done (in-order)
    } else {
      wait_async<0>();
    }
    __syncthreads();

    const unsigned char* la = stageA(kt & 1);
    const unsigned char* lb = stageB(kt & 1);

    bf16x16 bfrag[4];
    #pragma unroll
    for (int nt = 0; nt < 4; ++nt) {              // B: lane = column
      int n = wn * 64 + nt * 16 + (lane & 15);
      const unsigned char* p = lb + n * ROWB + (hi ? 32 : 0);
      bfrag[nt] = ld_frag(p, p + 16);
    }
    #pragma unroll
    for (int mt = 0; mt < 4; ++mt) {              // A: lane = row
      int r = wm * 64 + mt * 16 + (lane & 15);
      const unsigned char* p = la + r * ROWB + (hi ? 32 : 0);
      bf16x16 afrag = ld_frag(p, p + 16);
      #pragma unroll
      for (int nt = 0; nt < 4; ++nt)
        acc[mt][nt] = __builtin_amdgcn_wmma_f32_16x16x32_bf16(
            false, afrag, false, bfrag[nt], (short)0, acc[mt][nt], false, false);
    }
    __syncthreads();                              // stage reusable next iter
  }

  // epilogue: VGPR r -> row (r + hi*8), lane&15 -> col
  #pragma unroll
  for (int mt = 0; mt < 4; ++mt) {
    #pragma unroll
    for (int nt = 0; nt < 4; ++nt) {
      int col = n0 + wn * 64 + nt * 16 + (lane & 15);
      if (col >= N) continue;
      float bv = bias ? bias[col] : 0.0f;
      #pragma unroll
      for (int r = 0; r < 8; ++r) {
        int row = m0 + wm * 64 + mt * 16 + r + (hi ? 8 : 0);
        float v = acc[mt][nt][r] + bv;
        size_t idx = (size_t)row * N + col;
        if (EPI == EPI_EXPW) {
          float var = __expf(v);
          float wv  = 1.0f + sqrtf(var) * eps[idx];
          Cf[idx]  = var;
          C2f[idx] = wv;
          Cb[idx]  = f2bf(wv);
        } else {
          if (EPI == EPI_RELU) v = v > 0.f ? v : 0.f;
          if (Cf) Cf[idx] = v;
          if (Cb) Cb[idx] = f2bf(v);
        }
      }
    }
  }
}

// ---------------------------------------------------------------------------
// fp32 -> bf16 bulk convert (one pass per operand)
// ---------------------------------------------------------------------------
__global__ __launch_bounds__(256)
void cvt_bf16_kernel(const float* __restrict__ src, unsigned* __restrict__ dst, long n4)
{
  long i = (long)blockIdx.x * blockDim.x + threadIdx.x;
  if (i >= n4) return;
  f32x4 v = *(const f32x4*)(src + i * 4);
  dst[i * 2]     = pack2(v.x, v.y);
  dst[i * 2 + 1] = pack2(v.z, v.w);
}

// ---------------------------------------------------------------------------
// Stable two-bucket sort by mask = odd(mm): block-wide prefix scan.
// ---------------------------------------------------------------------------
__global__ __launch_bounds__(1024)
void perm_kernel(const int* __restrict__ mm, int Bn,
                 int* __restrict__ perm, int* __restrict__ pos)
{
  __shared__ int ssum[1024];
  const int t = threadIdx.x;
  const int PER = 4;
  int base = t * PER;
  int mk[PER]; int cnt = 0;
  #pragma unroll
  for (int i = 0; i < PER; ++i) {
    int idx = base + i;
    mk[i] = (idx < Bn) ? (mm[idx] & 1) : 0;
    cnt += mk[i];
  }
  ssum[t] = cnt;
  __syncthreads();
  for (int o = 1; o < 1024; o <<= 1) {
    int v = ssum[t];
    int u = (t >= o) ? ssum[t - o] : 0;
    __syncthreads();
    ssum[t] = v + u;
    __syncthreads();
  }
  int total_ones  = ssum[1023];
  int zeros_total = Bn - total_ones;
  int run = ssum[t] - cnt;
  #pragma unroll
  for (int i = 0; i < PER; ++i) {
    int idx = base + i;
    if (idx >= Bn) break;
    int dst = mk[i] ? (zeros_total + run) : (idx - run);
    perm[dst] = idx;
    pos[idx]  = dst;
    run += mk[i];
  }
}

// ---------------------------------------------------------------------------
// combined[pos[i],:] (bf16) = mask ? x_rec*softplus(mean + exp(0.5 l)*eps) : x
// ---------------------------------------------------------------------------
__global__ __launch_bounds__(256)
void combine_kernel(const float* __restrict__ x, const int* __restrict__ mm,
                    const int* __restrict__ pos,
                    const float* __restrict__ xr, const float* __restrict__ mreg,
                    const float* __restrict__ lreg, const float* __restrict__ eps_r,
                    ushort_t* __restrict__ cb, int Dd)
{
  int row = blockIdx.y;
  int j = blockIdx.x * blockDim.x + threadIdx.x;
  if (j >= Dd) return;
  size_t idx = (size_t)row * Dd + j;
  float val;
  if (mm[row] & 1) {
    float s   = __expf(0.5f * lreg[idx]);
    float reg = mreg[idx] + s * eps_r[idx];
    float sp  = (reg > 20.0f) ? reg : log1pf(__expf(reg));
    val = xr[idx] * sp;
  } else {
    val = x[idx];
  }
  cb[(size_t)pos[row] * Dd + j] = f2bf(val);
}

// ---------------------------------------------------------------------------
// out_w / out_var = w[perm[B/2:]], var[perm[B/2:]]
// ---------------------------------------------------------------------------
__global__ __launch_bounds__(256)
void gather_kernel(const int* __restrict__ perm,
                   const float* __restrict__ wv, const float* __restrict__ var,
                   float* __restrict__ out_w, float* __restrict__ out_var,
                   int Bn, int Pp)
{
  size_t i = (size_t)blockIdx.x * blockDim.x + threadIdx.x;
  size_t total = (size_t)(Bn / 2) * Pp;
  if (i >= total) return;
  int r = (int)(i / Pp), j = (int)(i % Pp);
  int src = perm[Bn / 2 + r];
  out_w[i]   = wv[(size_t)src * Pp + j];
  out_var[i] = var[(size_t)src * Pp + j];
}

// ---------------------------------------------------------------------------
// Host orchestration
// ---------------------------------------------------------------------------
extern "C" void kernel_launch(void* const* d_in, const int* in_sizes, int n_in,
                              void* d_out, int out_size, void* d_ws, size_t ws_size,
                              hipStream_t stream)
{
  (void)in_sizes; (void)n_in; (void)out_size; (void)ws_size;
  constexpr int Bn = 4096, Dd = 4096, Hh = 4096, Oo = 1000, Pp = 2048;

  const float* x    = (const float*)d_in[0];
  const int*   mm   = (const int*)  d_in[1];
  const float* pri  = (const float*)d_in[2];
  const float* W1   = (const float*)d_in[3];
  const float* b1   = (const float*)d_in[4];
  const float* W2   = (const float*)d_in[5];
  const float* b2   = (const float*)d_in[6];
  const float* Wr1  = (const float*)d_in[7];
  const float* br1  = (const float*)d_in[8];
  const float* Wlv  = (const float*)d_in[9];
  const float* blv  = (const float*)d_in[10];
  const float* Wg1  = (const float*)d_in[11];
  const float* bg1  = (const float*)d_in[12];
  const float* Wm   = (const float*)d_in[13];
  const float* bm   = (const float*)d_in[14];
  const float* Wl   = (const float*)d_in[15];
  const float* bl   = (const float*)d_in[16];
  const float* epsw = (const float*)d_in[17];
  const float* epsr = (const float*)d_in[18];

  float* out     = (float*)d_out;
  float* logits  = out;
  float* out_w   = out + (size_t)Bn * Oo;
  float* out_var = out_w + (size_t)(Bn / 2) * Pp;

  // fp32 workspace
  float* ws = (float*)d_ws;
  size_t off = 0;
  float* f_var = ws + off; off += (size_t)Bn * Pp;
  float* f_w   = ws + off; off += (size_t)Bn * Pp;
  float* f_xr  = ws + off; off += (size_t)Bn * Dd;
  float* f_m   = ws + off; off += (size_t)Bn * Dd;
  float* f_l   = ws + off; off += (size_t)Bn * Dd;
  int* perm = (int*)(ws + off);
  int* pos  = perm + Bn;

  // bf16 workspace (16B-aligned: all preceding sizes are multiples of 16B)
  ushort_t* bws = (ushort_t*)(pos + Bn);
  size_t boff = 0;
  ushort_t* bx   = bws + boff; boff += (size_t)Bn * Dd;   // x, reused for combined
  ushort_t* bWr1 = bws + boff; boff += (size_t)Hh * Dd;
  ushort_t* bWlv = bws + boff; boff += (size_t)Pp * Hh;
  ushort_t* bpri = bws + boff; boff += (size_t)Pp * Dd;
  ushort_t* bWg1 = bws + boff; boff += (size_t)Hh * Dd;
  ushort_t* bWm  = bws + boff; boff += (size_t)Dd * Hh;
  ushort_t* bWl  = bws + boff; boff += (size_t)Dd * Hh;
  ushort_t* bW1  = bws + boff; boff += (size_t)Hh * Dd;
  ushort_t* bW2  = bws + boff; boff += (size_t)Oo * Hh;
  ushort_t* bh   = bws + boff; boff += (size_t)Bn * Hh;   // h, reused for hr
  ushort_t* bw   = bws + boff; boff += (size_t)Bn * Pp;
  ushort_t* bxr  = bws + boff; boff += (size_t)Bn * Dd;   // x_rec, reused for h2

  dim3 blk(256);
  auto grid = [](int Mv, int Nv) { return dim3((Nv + BNT - 1) / BNT, (Mv + BMT - 1) / BMT); };
  auto cvt = [&](const float* s, ushort_t* d, size_t n) {
    cvt_bf16_kernel<<<(unsigned)((n / 4 + 255) / 256), blk, 0, stream>>>(s, (unsigned*)d, (long)(n / 4));
  };

  perm_kernel<<<1, 1024, 0, stream>>>(mm, Bn, perm, pos);

  // one-time fp32 -> bf16 conversions
  cvt(x,   bx,   (size_t)Bn * Dd);
  cvt(Wr1, bWr1, (size_t)Hh * Dd);
  cvt(Wlv, bWlv, (size_t)Pp * Hh);
  cvt(pri, bpri, (size_t)Pp * Dd);
  cvt(Wg1, bWg1, (size_t)Hh * Dd);
  cvt(Wm,  bWm,  (size_t)Dd * Hh);
  cvt(Wl,  bWl,  (size_t)Dd * Hh);
  cvt(W1,  bW1,  (size_t)Hh * Dd);
  cvt(W2,  bW2,  (size_t)Oo * Hh);

  // h = relu(x @ Wr1.T + br1)                      (bf16 out only)
  gemm_bf16<EPI_RELU, true ><<<grid(Bn, Hh), blk, 0, stream>>>(bx,  bWr1, br1, nullptr, nullptr, nullptr, bh,  Bn, Hh, Dd);
  // var = exp(h @ Wlv.T + blv); w = 1+sqrt(var)*eps_w
  gemm_bf16<EPI_EXPW, true ><<<grid(Bn, Pp), blk, 0, stream>>>(bh,  bWlv, blv, epsw,    f_var,  f_w,     bw,  Bn, Pp, Hh);
  // x_rec = w @ priors                             (fp32 + bf16 out)
  gemm_bf16<EPI_NONE, false><<<grid(Bn, Dd), blk, 0, stream>>>(bw,  bpri, nullptr, nullptr, f_xr, nullptr, bxr, Bn, Dd, Pp);
  // hr = relu(x_rec @ Wg1.T + bg1)                 (bf16 out only)
  gemm_bf16<EPI_RELU, true ><<<grid(Bn, Hh), blk, 0, stream>>>(bxr, bWg1, bg1, nullptr, nullptr, nullptr, bh,  Bn, Hh, Dd);
  // mean_reg / raw-logstd (fp32 only)
  gemm_bf16<EPI_NONE, true ><<<grid(Bn, Dd), blk, 0, stream>>>(bh,  bWm,  bm,  nullptr, f_m,    nullptr, nullptr, Bn, Dd, Hh);
  gemm_bf16<EPI_NONE, true ><<<grid(Bn, Dd), blk, 0, stream>>>(bh,  bWl,  bl,  nullptr, f_l,    nullptr, nullptr, Bn, Dd, Hh);

  combine_kernel<<<dim3((Dd + 255) / 256, Bn), blk, 0, stream>>>(x, mm, pos, f_xr, f_m, f_l, epsr, bx, Dd);

  // logits = relu(comb @ W1.T + b1) @ W2.T + b2
  gemm_bf16<EPI_RELU, true ><<<grid(Bn, Hh), blk, 0, stream>>>(bx,  bW1, b1, nullptr, nullptr, nullptr, bxr, Bn, Hh, Dd);
  gemm_bf16<EPI_NONE, true ><<<grid(Bn, Oo), blk, 0, stream>>>(bxr, bW2, b2, nullptr, logits,  nullptr, nullptr, Bn, Oo, Hh);

  gather_kernel<<<(unsigned)(((size_t)(Bn / 2) * Pp + 255) / 256), blk, 0, stream>>>(
      perm, f_w, f_var, out_w, out_var, Bn, Pp);
}
